// PredictionNet_21114059227170
// MI455X (gfx1250) — compile-verified, matrix operand order checked by reading, not run on hardware
//
#include <hip/hip_runtime.h>
#include <stdint.h>

// ---- trivial ext-vector types (safe in unions) ----
typedef __attribute__((ext_vector_type(16))) __bf16   v16bf;
typedef __attribute__((ext_vector_type(8)))  float    v8f;
typedef __attribute__((ext_vector_type(4)))  float    f32x4;
typedef __attribute__((ext_vector_type(4)))  unsigned u32x4;

#define B_     128
#define E_     6
#define IN_    1664
#define H_     512
#define OUT_   618
#define ZD_    32
#define K2_    (H_ + ZD_)   // 544
#define OPAD3_ 624          // 618 padded to 16-multiple (39 tiles)

__device__ __forceinline__ uint16_t f2bf(float f) {
  uint32_t u = __float_as_uint(f);
  u += 0x7FFFu + ((u >> 16) & 1u);      // round-to-nearest-even
  return (uint16_t)(u >> 16);
}

union Frag  { v16bf v; uint16_t s[16]; u32x4 q[2]; };
union Acc   { v8f v; float f[8]; };
union Pack8 { u32x4 q; uint16_t s[8]; };

// ---------------------------------------------------------------------------
// Weight prep: fp32 [E][K][O] -> bf16 [E][Opad][K] (transposed, zero-padded)
// ---------------------------------------------------------------------------
__global__ __launch_bounds__(256)
void convert_transpose(const float* __restrict__ w, uint16_t* __restrict__ wt,
                       int K, int O, int Opad) {
  __shared__ float tile[32][33];
  const int tx = threadIdx.x, ty = threadIdx.y;
  const int k0 = blockIdx.x * 32, o0 = blockIdx.y * 32, e = blockIdx.z;
  const float* wb = w + (size_t)e * K * O;
  #pragma unroll
  for (int i = 0; i < 4; ++i) {
    int ky = ty + 8 * i, o = o0 + tx;
    tile[ky][tx] = (o < O) ? wb[(size_t)(k0 + ky) * O + o] : 0.0f;
  }
  __syncthreads();
  uint16_t* wo = wt + (size_t)e * Opad * K;
  #pragma unroll
  for (int i = 0; i < 4; ++i) {
    int oy = ty + 8 * i;
    wo[(size_t)(o0 + oy) * K + (k0 + tx)] = f2bf(tile[tx][oy]);
  }
}

// coefA[b][j] = bf16(coef[b,j]) for j<6 else 0   (shared by all 3 layers)
__global__ void build_coefA(const float* __restrict__ coef, uint16_t* __restrict__ coefA) {
  int idx = blockIdx.x * blockDim.x + threadIdx.x;   // 128*32
  int b = idx >> 5, j = idx & 31;
  coefA[idx] = (j < E_) ? f2bf(coef[b * E_ + j]) : (uint16_t)0;
}

// biasT[o][j] = bf16(bias[j,o]) for j<6 else 0 ; o in [0,Opad)
__global__ void build_biasT(const float* __restrict__ bias, uint16_t* __restrict__ biasT,
                            int O, int Opad) {
  int idx = blockIdx.x * blockDim.x + threadIdx.x;
  if (idx >= Opad * 32) return;
  int o = idx >> 5, j = idx & 31;
  float v = (j < E_ && o < O) ? bias[j * O + o] : 0.0f;
  biasT[idx] = f2bf(v);
}

// copy z into cols [512,544) of both intermediate activation buffers
__global__ void copy_z(const float* __restrict__ z,
                       float* __restrict__ x1, float* __restrict__ x2) {
  int idx = blockIdx.x * blockDim.x + threadIdx.x;   // 128*32
  int b = idx >> 5, j = idx & 31;
  float v = z[(size_t)b * ZD_ + j];
  x1[(size_t)b * K2_ + H_ + j] = v;
  x2[(size_t)b * K2_ + H_ + j] = v;
}

// Abf[e][b][k] = bf16(coef[b,e] * X[b,k]) ; 8 k-elements per thread
__global__ __launch_bounds__(256)
void prescale_A(const float* __restrict__ X, const float* __restrict__ coef,
                uint16_t* __restrict__ Abf, int K, int total8) {
  int idx = blockIdx.x * blockDim.x + threadIdx.x;
  if (idx >= total8) return;
  int k8  = (idx % (K >> 3)) << 3;
  int tmp = idx / (K >> 3);
  int b = tmp % B_, e = tmp / B_;
  const float c = coef[b * E_ + e];
  const float* xr = X + (size_t)b * K + k8;
  f32x4 v0 = *(const f32x4*)(xr);
  f32x4 v1 = *(const f32x4*)(xr + 4);
  Pack8 p;
  p.s[0] = f2bf(c * v0.x); p.s[1] = f2bf(c * v0.y);
  p.s[2] = f2bf(c * v0.z); p.s[3] = f2bf(c * v0.w);
  p.s[4] = f2bf(c * v1.x); p.s[5] = f2bf(c * v1.y);
  p.s[6] = f2bf(c * v1.z); p.s[7] = f2bf(c * v1.w);
  *(u32x4*)(Abf + ((size_t)(e * B_ + b)) * K + k8) = p.q;
}

// ---------------------------------------------------------------------------
// WMMA GEMM: acc[b,o] = sum_e sum_k Abf[e,b,k]*Wt[e,o,k]  (+ coefA x biasT)
// Block: 128 threads = 4 waves; wave owns 16 rows; gridDim.y splits M (2x64).
// ---------------------------------------------------------------------------
template<int NT, bool ELU>
__global__ __launch_bounds__(128)
void moe_wmma_gemm(const uint16_t* __restrict__ Abf, int K,
                   const uint16_t* __restrict__ coefA,
                   const uint16_t* __restrict__ biasT,
                   const uint16_t* __restrict__ Wt, int Opad, int Ocols,
                   float* __restrict__ Y, int sy) {
  const int lane = threadIdx.x & 31;
  const int wave = threadIdx.x >> 5;                 // 0..3
  const int m    = lane & 15;
  const int hi   = (lane >> 4) & 1;
  const int hi8  = hi * 8;
  const int row0 = (blockIdx.y * 4 + wave) * 16;
  const int row  = row0 + m;
  const int nb   = blockIdx.x * (NT * 16);

  Acc acc[NT];
  #pragma unroll
  for (int t = 0; t < NT; ++t)
    #pragma unroll
    for (int r = 0; r < 8; ++r) acc[t].f[r] = 0.0f;

  for (int e = 0; e < E_; ++e) {
    const uint16_t* ar = Abf + ((size_t)(e * B_ + row)) * K;
    const uint16_t* wb = Wt + (size_t)e * Opad * K;
    for (int kb = 0; kb < K; kb += 32) {
      Frag a;
      a.q[0] = *(const u32x4*)(ar + kb + hi8);        // K = hi8..hi8+7
      a.q[1] = *(const u32x4*)(ar + kb + 16 + hi8);   // K = 16+hi8..+7
      #pragma unroll
      for (int t = 0; t < NT; ++t) {
        const uint16_t* wr = wb + (size_t)(nb + 16 * t + m) * K + kb;
        Frag b;
        b.q[0] = *(const u32x4*)(wr + hi8);
        b.q[1] = *(const u32x4*)(wr + 16 + hi8);
        acc[t].v = __builtin_amdgcn_wmma_f32_16x16x32_bf16(
            false, a.v, false, b.v, (short)0, acc[t].v, false, false);
      }
    }
  }

  // blended bias as one extra WMMA K-block: A = coefA row, B = biasT rows
  {
    const uint16_t* cr = coefA + (size_t)row * 32;
    Frag a;
    a.q[0] = *(const u32x4*)(cr + hi8);
    a.q[1] = *(const u32x4*)(cr + 16 + hi8);
    #pragma unroll
    for (int t = 0; t < NT; ++t) {
      const uint16_t* br = biasT + (size_t)(nb + 16 * t + m) * 32;
      Frag b;
      b.q[0] = *(const u32x4*)(br + hi8);
      b.q[1] = *(const u32x4*)(br + 16 + hi8);
      acc[t].v = __builtin_amdgcn_wmma_f32_16x16x32_bf16(
          false, a.v, false, b.v, (short)0, acc[t].v, false, false);
    }
  }

  // epilogue: optional ELU, write fp32
  #pragma unroll
  for (int t = 0; t < NT; ++t) {
    const int col = nb + 16 * t + m;
    if (col >= Ocols) continue;
    #pragma unroll
    for (int r = 0; r < 8; ++r) {
      const int orow = row0 + hi * 8 + r;            // C layout: VGPR r -> M=r(+8)
      float v = acc[t].f[r];
      if (ELU) v = (v > 0.0f) ? v : (__expf(v) - 1.0f);
      Y[(size_t)orow * sy + col] = v;
    }
  }
}

// ---------------------------------------------------------------------------
extern "C" void kernel_launch(void* const* d_in, const int* in_sizes, int n_in,
                              void* d_out, int out_size, void* d_ws, size_t ws_size,
                              hipStream_t stream) {
  (void)in_sizes; (void)n_in; (void)out_size; (void)ws_size;
  const float* p_prev = (const float*)d_in[0];
  const float* coef   = (const float*)d_in[1];
  const float* z      = (const float*)d_in[2];
  const float* w1 = (const float*)d_in[3];
  const float* b1 = (const float*)d_in[4];
  const float* w2 = (const float*)d_in[5];
  const float* b2 = (const float*)d_in[6];
  const float* w3 = (const float*)d_in[7];
  const float* b3 = (const float*)d_in[8];
  float* out = (float*)d_out;

  // workspace carve-up (~21 MB total)
  char* ws = (char*)d_ws;
  size_t off = 0;
  auto carve = [&](size_t bytes) {
    void* p = ws + off;
    off += (bytes + 255) & ~(size_t)255;
    return p;
  };
  uint16_t* wt1   = (uint16_t*)carve((size_t)E_ * H_ * IN_ * 2);      // [6][512][1664]
  uint16_t* wt2   = (uint16_t*)carve((size_t)E_ * H_ * K2_ * 2);      // [6][512][544]
  uint16_t* wt3   = (uint16_t*)carve((size_t)E_ * OPAD3_ * K2_ * 2);  // [6][624][544]
  uint16_t* bt1   = (uint16_t*)carve((size_t)H_ * 32 * 2);            // [512][32]
  uint16_t* bt2   = (uint16_t*)carve((size_t)H_ * 32 * 2);
  uint16_t* bt3   = (uint16_t*)carve((size_t)OPAD3_ * 32 * 2);        // [624][32]
  uint16_t* coefA = (uint16_t*)carve((size_t)B_ * 32 * 2);            // [128][32]
  float*    x1    = (float*)carve((size_t)B_ * K2_ * 4);              // [128][544]
  float*    x2    = (float*)carve((size_t)B_ * K2_ * 4);
  uint16_t* abf   = (uint16_t*)carve((size_t)E_ * B_ * IN_ * 2);      // reused all layers

  dim3 tb(32, 8, 1);
  convert_transpose<<<dim3(IN_ / 32, H_ / 32, E_),     tb, 0, stream>>>(w1, wt1, IN_, H_,   H_);
  convert_transpose<<<dim3(K2_ / 32, H_ / 32, E_),     tb, 0, stream>>>(w2, wt2, K2_, H_,   H_);
  convert_transpose<<<dim3(K2_ / 32, OPAD3_ / 32, E_), tb, 0, stream>>>(w3, wt3, K2_, OUT_, OPAD3_);
  build_coefA<<<(B_ * 32) / 256, 256, 0, stream>>>(coef, coefA);
  build_biasT<<<(H_ * 32) / 256, 256, 0, stream>>>(b1, bt1, H_, H_);
  build_biasT<<<(H_ * 32) / 256, 256, 0, stream>>>(b2, bt2, H_, H_);
  build_biasT<<<(OPAD3_ * 32 + 255) / 256, 256, 0, stream>>>(b3, bt3, OUT_, OPAD3_);
  copy_z<<<(B_ * ZD_) / 256, 256, 0, stream>>>(z, x1, x2);

  const int t8_1 = E_ * B_ * (IN_ / 8);   // prescale work items, layer 1
  const int t8_2 = E_ * B_ * (K2_ / 8);   // layers 2/3

  // layer 1: [128,1664] x [6,1664,512] -> x1[:, :512], ELU
  prescale_A<<<(t8_1 + 255) / 256, 256, 0, stream>>>(p_prev, coef, abf, IN_, t8_1);
  moe_wmma_gemm<4, true ><<<dim3(H_ / 64, 2), 128, 0, stream>>>(
      abf, IN_, coefA, bt1, wt1, H_, H_, x1, K2_);

  // layer 2: [128,544] x [6,544,512] -> x2[:, :512], ELU
  prescale_A<<<(t8_2 + 255) / 256, 256, 0, stream>>>(x1, coef, abf, K2_, t8_2);
  moe_wmma_gemm<4, true ><<<dim3(H_ / 64, 2), 128, 0, stream>>>(
      abf, K2_, coefA, bt2, wt2, H_, H_, x2, K2_);

  // layer 3: [128,544] x [6,544,618] -> out, no act
  prescale_A<<<(t8_2 + 255) / 256, 256, 0, stream>>>(x2, coef, abf, K2_, t8_2);
  moe_wmma_gemm<1, false><<<dim3(OPAD3_ / 16, 2), 128, 0, stream>>>(
      abf, K2_, coefA, bt3, wt3, OPAD3_, OUT_, out, OUT_);
}
